// NeuralCellularAutomata_24610162606403
// MI455X (gfx1250) — matrix operand order repaired and checked by reading, not compiled, not run on hardware
//
#include <hip/hip_runtime.h>
#include <hip/hip_bf16.h>
#include <stdint.h>

typedef __attribute__((ext_vector_type(16))) _Float16 v16h;
typedef __attribute__((ext_vector_type(8)))  float    v8f;
typedef __attribute__((ext_vector_type(4)))  float    v4f;
typedef __attribute__((ext_vector_type(4)))  unsigned int v4u;
typedef __attribute__((ext_vector_type(2)))  __fp16   v2fp;

union Frag { v16h h; v4u q[2]; };
union F8   { v8f  f; v4f q[2]; };

static __device__ inline unsigned short f2h(float x) {
    _Float16 hf = (_Float16)x;
    union { _Float16 f; unsigned short u; } cv; cv.f = hf; return cv.u;
}
// packed f32x2 -> f16x2 in one v_cvt_pk_rtz_f16_f32
static __device__ inline unsigned int packrtz(float a, float b) {
    union { v2fp h; unsigned int u; } cv;
    cv.h = __builtin_amdgcn_cvt_pkrtz(a, b);
    return cv.u;
}
static __device__ inline float relu1(float x) {
    return __builtin_fmaxf(x, 0.f);
}

// ---------------------------------------------------------------------------
// Kernel 0: fold weights.
// W1 (128,48) f32 -> W1e f16 [o=128][k=32]: k 0..15 = W1[o][32+c] (x part),
// k16 = sum W1[o][0..15] (sx group), k17 = sum W1[o][16..31] (sy group), rest 0.
// b1eff = b1 + W1[:,:16]@bias_x + W1[:,16:32]@bias_y.   W2 -> f16 [o=16][k=128].
// ---------------------------------------------------------------------------
__global__ void nca_prep(const float* __restrict__ W1, const float* __restrict__ b1,
                         const float* __restrict__ bx, const float* __restrict__ by,
                         const float* __restrict__ W2,
                         unsigned short* __restrict__ w1e, float* __restrict__ b1e,
                         unsigned short* __restrict__ w2e) {
    int o = threadIdx.x;  // 0..127
    const float* r = W1 + o * 48;
    float ssum = 0.f, tsum = 0.f, be = b1[o];
    #pragma unroll
    for (int c = 0; c < 16; ++c) {
        ssum += r[c];
        tsum += r[16 + c];
        be += r[c] * bx[c] + r[16 + c] * by[c];
    }
    #pragma unroll
    for (int c = 0; c < 16; ++c) w1e[o * 32 + c] = f2h(r[32 + c]);
    w1e[o * 32 + 16] = f2h(ssum);
    w1e[o * 32 + 17] = f2h(tsum);
    #pragma unroll
    for (int c = 18; c < 32; ++c) w1e[o * 32 + c] = 0;
    b1e[o] = be;
    for (int i = o; i < 16 * 128; i += 128) w2e[i] = f2h(W2[i]);
}

// ---------------------------------------------------------------------------
// Fused kernel: one pass over x.
// Block (256 thr, 8 waves) owns a 32x32 pixel tile of one batch image.
// Phase A: cooperative load of x -> LDS: 34x34 halo of channel sums (f32)
//          + 32x32 packed-f16 channel pairs (row stride 260 uints, no bank
//          conflicts for the per-lane fragment loads).
// Phase B: per wave, 8 subtiles of 16 pixels along h (N axis = h so the
//          output stores are 1KB-contiguous):
//   GEMM1: Ht(16f x 16pix) = W1e_t(16x32) x Pt^T(32x16)   (8 WMMA)
//   GEMM2: out^T(16o x 16pix) = W2(16x128) x H^T(128x16)  (4 chained WMMA)
// ---------------------------------------------------------------------------
__global__ __launch_bounds__(256) void nca_fused(
    const float* __restrict__ x,
    const unsigned short* __restrict__ w1e, const float* __restrict__ b1e,
    const unsigned short* __restrict__ w2e, const float* __restrict__ b2g,
    float* __restrict__ out) {
    __shared__ __align__(16) unsigned int xt[32 * 260];   // [hrel][wrel*8+cp], padded
    __shared__ __align__(16) float ssum[34 * 34];         // halo channel sums
    __shared__ __align__(16) unsigned int ldsH[8][16 * 64]; // per-wave [pixel][feat-pair]
    __shared__ __align__(16) float sb1[128];

    const int tid = threadIdx.x, wid = tid >> 5, lane = tid & 31;
    unsigned int* lH = ldsH[wid];
    const int col = lane & 15;     // B/N axis: pixel (h); A/M axis row index
    const int hi  = lane >> 4;     // K-half select

    const int blk = blockIdx.x;    // 32 b x 4 th x 4 tw = 512
    const int b  = blk >> 4;
    const int h0 = ((blk >> 2) & 3) * 32;
    const int w0 = (blk & 3) * 32;

    if (tid < 128) sb1[tid] = b1e[tid];

    // ---- loop-invariant A fragments (weights) ----
    // A-frag pairs: lanes<16 -> {0..7}; lanes>=16 -> {8..15}
    Frag A1[8], A2[4];
    const v4u* w1q = (const v4u*)w1e;
    const v4u* w2q = (const v4u*)w2e;
    #pragma unroll
    for (int t = 0; t < 8; ++t) {
        int base = (16 * t + col) * 4 + 2 * hi;   // uint4 units, row = feature 16t+col
        A1[t].q[0] = w1q[base];
        A1[t].q[1] = w1q[base + 1];
    }
    #pragma unroll
    for (int k = 0; k < 4; ++k) {
        int base = col * 16 + 4 * k + 2 * hi;     // row = output o = col
        A2[k].q[0] = w2q[base];
        A2[k].q[1] = w2q[base + 1];
    }
    F8 bias2F;                                 // C frag for first GEMM2 step
    #pragma unroll
    for (int j = 0; j < 8; ++j) bias2F.f[j] = b2g[8 * hi + j];

    // ---- Phase A: stage sums (halo) + packed x (interior) ----
    for (int i = tid; i < 34 * 34; i += 256) {
        int hr = i / 34;            // 0..33 ; global h = h0 + hr - 1
        int wr = i - hr * 34;       // 0..33 ; global w = w0 + wr - 1
        int hg = h0 + hr - 1, wg = w0 + wr - 1;
        float s = 0.f;
        if (hg >= 0 && hg < 128 && wg >= 0 && wg < 128) {
            const float* p = x + b * (16 * 16384) + hg * 128 + wg;
            float v[16];
            #pragma unroll
            for (int c = 0; c < 16; ++c) { v[c] = p[c * 16384]; s += v[c]; }
            if ((unsigned)(hr - 1) < 32u && (unsigned)(wr - 1) < 32u) {
                unsigned int* xp = xt + (hr - 1) * 260 + (wr - 1) * 8;
                v4u q0, q1;
                q0.x = packrtz(v[0],  v[1]);  q0.y = packrtz(v[2],  v[3]);
                q0.z = packrtz(v[4],  v[5]);  q0.w = packrtz(v[6],  v[7]);
                q1.x = packrtz(v[8],  v[9]);  q1.y = packrtz(v[10], v[11]);
                q1.z = packrtz(v[12], v[13]); q1.w = packrtz(v[14], v[15]);
                *(v4u*)xp       = q0;
                *(v4u*)(xp + 4) = q1;
            }
        }
        ssum[i] = s;
    }
    __syncthreads();

    // ---- Phase B: 8 subtiles per wave: ww = wid*4 + (st>>1), h-half = st&1 ----
    for (int st = 0; st < 8; ++st) {
        int ww = wid * 4 + (st >> 1);   // 0..31, column within tile
        int sh = st & 1;                // h half: pixels h0 + sh*16 + 0..15

        // ---- B fragment of P^T, built directly in registers ----
        Frag B1;
        if (lane < 16) {
            // K-pairs 0..7 = x channels of pixel (h = sh*16+col, w = ww)
            const v4u* xq = (const v4u*)(xt + (sh * 16 + col) * 260 + ww * 8);
            B1.q[0] = xq[0];
            B1.q[1] = xq[1];
        } else {
            // K-pairs 8..15 = { (s,t), zeros } ; sobel from halo sums
            int hrel = sh * 16 + col;
            const float* s0 = ssum + hrel * 34 + ww;
            float a00 = s0[0],  a01 = s0[1],  a02 = s0[2];
            float a10 = s0[34],               a12 = s0[36];
            float a20 = s0[68], a21 = s0[69], a22 = s0[70];
            float sv = (a02 - a00 + 2.f * (a12 - a10) + a22 - a20) * 0.125f;
            float tv = (a20 - a00 + 2.f * (a21 - a01) + a22 - a02) * 0.125f;
            v4u qs; qs.x = packrtz(sv, tv); qs.y = 0u; qs.z = 0u; qs.w = 0u;
            B1.q[0] = qs;
            v4u zz; zz.x = 0u; zz.y = 0u; zz.z = 0u; zz.w = 0u;
            B1.q[1] = zz;
        }

        // ---- GEMM1: 8 feature tiles; D rows are consecutive features ----
        #pragma unroll
        for (int t = 0; t < 8; ++t) {
            F8 cb;   // bias broadcast: acc[j] = b1eff[16t + 8*hi + j]
            const v4f* bq = (const v4f*)(sb1 + 16 * t + 8 * hi);
            cb.q[0] = bq[0];
            cb.q[1] = bq[1];
            v8f acc = __builtin_amdgcn_wmma_f32_16x16x32_f16(
                false, A1[t].h, false, B1.h, (short)0, cb.f, false, false);
            v4u hq;   // relu + packed convert: features 16t+8hi+2j, +1
            hq.x = packrtz(relu1(acc[0]), relu1(acc[1]));
            hq.y = packrtz(relu1(acc[2]), relu1(acc[3]));
            hq.z = packrtz(relu1(acc[4]), relu1(acc[5]));
            hq.w = packrtz(relu1(acc[6]), relu1(acc[7]));
            *(v4u*)(lH + col * 64 + 8 * t + 4 * hi) = hq;
        }
        asm volatile("s_wait_dscnt 0x0" ::: "memory");

        // ---- GEMM2: 4 chained K-steps, bias as C of the first ----
        v8f acc2;
        {
            Frag Bh;
            const v4u* hq = (const v4u*)(lH + col * 64 + hi * 8);
            Bh.q[0] = hq[0];
            Bh.q[1] = hq[1];
            acc2 = __builtin_amdgcn_wmma_f32_16x16x32_f16(
                false, A2[0].h, false, Bh.h, (short)0, bias2F.f, false, false);
        }
        #pragma unroll
        for (int k = 1; k < 4; ++k) {
            Frag Bh;
            const v4u* hq = (const v4u*)(lH + col * 64 + 16 * k + hi * 8);
            Bh.q[0] = hq[0];
            Bh.q[1] = hq[1];
            acc2 = __builtin_amdgcn_wmma_f32_16x16x32_f16(
                false, A2[k].h, false, Bh.h, (short)0, acc2, false, false);
        }

        // ---- store: lane = pixel h, 8 consecutive o -> contiguous 1KB/subtile
        F8 r; r.f = acc2;
        float* op = out + ((b * 128 + w0 + ww) * 128 + h0 + sh * 16 + col) * 16 + 8 * hi;
        *(v4f*)op = r.q[0];
        *((v4f*)op + 1) = r.q[1];
    }
}

// ---------------------------------------------------------------------------
extern "C" void kernel_launch(void* const* d_in, const int* in_sizes, int n_in,
                              void* d_out, int out_size, void* d_ws, size_t ws_size,
                              hipStream_t stream) {
    const float* x  = (const float*)d_in[0];   // (32,16,128,128)
    const float* bx = (const float*)d_in[1];   // (16,)
    const float* by = (const float*)d_in[2];   // (16,)
    const float* W1 = (const float*)d_in[3];   // (128,48)
    const float* b1 = (const float*)d_in[4];   // (128,)
    const float* W2 = (const float*)d_in[5];   // (16,128)
    const float* b2 = (const float*)d_in[6];   // (16,)
    float* out = (float*)d_out;                // (32,128,128,16)

    char* ws = (char*)d_ws;
    unsigned short* w1e = (unsigned short*)ws;                  // 8 KB
    float*          b1e = (float*)(ws + 8 * 1024);              // 512 B
    unsigned short* w2e = (unsigned short*)(ws + 8 * 1024 + 512); // 4 KB

    nca_prep<<<1, 128, 0, stream>>>(W1, b1, bx, by, W2, w1e, b1e, w2e);
    nca_fused<<<512, 256, 0, stream>>>(x, w1e, b1e, w2e, b2, out);
}
